// Encoder_16389595201848
// MI455X (gfx1250) — matrix operand order repaired and checked by reading, not compiled, not run on hardware
//
#include <hip/hip_runtime.h>

typedef __attribute__((ext_vector_type(16))) _Float16 v16h;
typedef __attribute__((ext_vector_type(8)))  float    v8f;

#define NPAPER  50000
#define NAUTHOR 20000
#define DIM     128

// ----------------------------------------------------------------------------
// Phase 1: edge scatter. One wave (32 lanes) per edge; lane j handles 4 floats.
// Accumulators live in workspace (62 MB total -> resident in 192 MB L2, so the
// f32 atomics run at L2 rate; HBM only sees the 512B row gathers).
// ----------------------------------------------------------------------------
__global__ void scatter_accum(const float* __restrict__ xsrc,
                              const int* __restrict__ src,
                              const int* __restrict__ dst,
                              float* __restrict__ sums,
                              float* __restrict__ cnt,
                              int nedges) {
  int gwave = (int)((blockIdx.x * blockDim.x + threadIdx.x) >> 5);
  int lane  = threadIdx.x & 31;
  if (gwave >= nedges) return;
  int s = src[gwave];
  int d = dst[gwave];
  const float4 v = ((const float4*)(xsrc + (size_t)s * DIM))[lane];
  float* p = sums + (size_t)d * DIM + lane * 4;
  atomicAdd(p + 0, v.x);
  atomicAdd(p + 1, v.y);
  atomicAdd(p + 2, v.z);
  atomicAdd(p + 3, v.w);
  if (lane == 0) atomicAdd(cnt + d, 1.0f);
}

// ----------------------------------------------------------------------------
// WMMA fragment builders (wave32, 16x16x32 f16 -> f32).
// A (16x32, MxK): lane L holds row m=L&15; khalf=L>>4.
//   halves h=0..15 -> k = kb*32 + khalf*16 + h
// B (32x16, KxN): lane L holds col n=L&15; halves h=0..15 -> k = kb*32 + khalf*16 + h
// C/D (16x16 f32): lane L col n=L&15; vgpr r -> row m = (L>>4)*8 + r
// ----------------------------------------------------------------------------
__device__ __forceinline__ v16h make_a(const float* __restrict__ rowp,
                                       int kb, int khalf, float s) {
  const float4* p0 = (const float4*)(rowp + kb * 32 + khalf * 16);
  const float4* p1 = (const float4*)(rowp + kb * 32 + khalf * 16 + 8);
  float4 q0 = p0[0], q1 = p0[1], q2 = p1[0], q3 = p1[1];
  v16h a;
  a[0]  = (_Float16)(q0.x * s); a[1]  = (_Float16)(q0.y * s);
  a[2]  = (_Float16)(q0.z * s); a[3]  = (_Float16)(q0.w * s);
  a[4]  = (_Float16)(q1.x * s); a[5]  = (_Float16)(q1.y * s);
  a[6]  = (_Float16)(q1.z * s); a[7]  = (_Float16)(q1.w * s);
  a[8]  = (_Float16)(q2.x * s); a[9]  = (_Float16)(q2.y * s);
  a[10] = (_Float16)(q2.z * s); a[11] = (_Float16)(q2.w * s);
  a[12] = (_Float16)(q3.x * s); a[13] = (_Float16)(q3.y * s);
  a[14] = (_Float16)(q3.z * s); a[15] = (_Float16)(q3.w * s);
  return a;
}

__device__ __forceinline__ v16h make_b(const float* __restrict__ wrow,
                                       int kb, int khalf) {
  const float4* p = (const float4*)(wrow + kb * 32 + khalf * 16);
  float4 q0 = p[0], q1 = p[1], q2 = p[2], q3 = p[3];
  v16h b;
  b[0]  = (_Float16)q0.x; b[1]  = (_Float16)q0.y; b[2]  = (_Float16)q0.z; b[3]  = (_Float16)q0.w;
  b[4]  = (_Float16)q1.x; b[5]  = (_Float16)q1.y; b[6]  = (_Float16)q1.z; b[7]  = (_Float16)q1.w;
  b[8]  = (_Float16)q2.x; b[9]  = (_Float16)q2.y; b[10] = (_Float16)q2.z; b[11] = (_Float16)q2.w;
  b[12] = (_Float16)q3.x; b[13] = (_Float16)q3.y; b[14] = (_Float16)q3.z; b[15] = (_Float16)q3.w;
  return b;
}

__device__ __forceinline__ v16h make_b2(const float* __restrict__ wrow0,
                                        const float* __restrict__ wrow1,
                                        int kb, int khalf) {
  const float4* p = (const float4*)(wrow0 + kb * 32 + khalf * 16);
  const float4* q = (const float4*)(wrow1 + kb * 32 + khalf * 16);
  float4 a0 = p[0], a1 = p[1], a2 = p[2], a3 = p[3];
  float4 b0 = q[0], b1 = q[1], b2 = q[2], b3 = q[3];
  v16h b;
  b[0]  = (_Float16)(a0.x + b0.x); b[1]  = (_Float16)(a0.y + b0.y);
  b[2]  = (_Float16)(a0.z + b0.z); b[3]  = (_Float16)(a0.w + b0.w);
  b[4]  = (_Float16)(a1.x + b1.x); b[5]  = (_Float16)(a1.y + b1.y);
  b[6]  = (_Float16)(a1.z + b1.z); b[7]  = (_Float16)(a1.w + b1.w);
  b[8]  = (_Float16)(a2.x + b2.x); b[9]  = (_Float16)(a2.y + b2.y);
  b[10] = (_Float16)(a2.z + b2.z); b[11] = (_Float16)(a2.w + b2.w);
  b[12] = (_Float16)(a3.x + b3.x); b[13] = (_Float16)(a3.y + b3.y);
  b[14] = (_Float16)(a3.z + b3.z); b[15] = (_Float16)(a3.w + b3.w);
  return b;
}

#define WMMA_F16(A, B, C) \
  __builtin_amdgcn_wmma_f32_16x16x32_f16(false, (A), false, (B), (short)0, (C), false, false)

// ----------------------------------------------------------------------------
// Phase 2a: paper output = mean_w @ Wr_w^T + mean_c @ Wr_c^T
//                        + x_paper @ (Wo_w + Wo_c)^T + (br_w + br_c)
// One wave per 16x16 output tile. 3125 row-tiles x 8 col-tiles = 25000 waves.
// ----------------------------------------------------------------------------
__global__ void paper_out_kernel(const float* __restrict__ sums_w,
                                 const float* __restrict__ cnt_w,
                                 const float* __restrict__ sums_c,
                                 const float* __restrict__ cnt_c,
                                 const float* __restrict__ x_paper,
                                 const float* __restrict__ Wr_w,
                                 const float* __restrict__ br_w,
                                 const float* __restrict__ Wr_c,
                                 const float* __restrict__ br_c,
                                 const float* __restrict__ Wo_w,
                                 const float* __restrict__ Wo_c,
                                 float* __restrict__ out) {
  int waveId = threadIdx.x >> 5;
  int lane   = threadIdx.x & 31;
  int tile   = blockIdx.x * (blockDim.x >> 5) + waveId;
  int mt = tile >> 3;          // row tile (0..3124)
  int nt = tile & 7;           // col tile (0..7)
  int m     = mt * 16 + (lane & 15);
  int n     = nt * 16 + (lane & 15);
  int khalf = lane >> 4;

  float inv_w = 1.0f / fmaxf(cnt_w[m], 1.0f);
  float inv_c = 1.0f / fmaxf(cnt_c[m], 1.0f);
  const float* rw = sums_w  + (size_t)m * DIM;
  const float* rc = sums_c  + (size_t)m * DIM;
  const float* rx = x_paper + (size_t)m * DIM;
  const float* ww = Wr_w + (size_t)n * DIM;
  const float* wc = Wr_c + (size_t)n * DIM;
  const float* wo0 = Wo_w + (size_t)n * DIM;
  const float* wo1 = Wo_c + (size_t)n * DIM;

  v8f acc = {};
#pragma unroll
  for (int kb = 0; kb < 4; ++kb) {
    acc = WMMA_F16(make_a(rw, kb, khalf, inv_w), make_b(ww, kb, khalf), acc);
    acc = WMMA_F16(make_a(rc, kb, khalf, inv_c), make_b(wc, kb, khalf), acc);
    acc = WMMA_F16(make_a(rx, kb, khalf, 1.0f),  make_b2(wo0, wo1, kb, khalf), acc);
  }

  float bias = br_w[n] + br_c[n];
  int mrow = mt * 16 + khalf * 8;
#pragma unroll
  for (int r = 0; r < 8; ++r)
    out[(size_t)(mrow + r) * DIM + n] = acc[r] + bias;
}

// ----------------------------------------------------------------------------
// Phase 2b: author output = mean_a @ Wr^T + x_author @ Wo^T + br
// 1250 row-tiles x 8 col-tiles = 10000 waves.
// ----------------------------------------------------------------------------
__global__ void author_out_kernel(const float* __restrict__ sums_a,
                                  const float* __restrict__ cnt_a,
                                  const float* __restrict__ x_author,
                                  const float* __restrict__ Wr,
                                  const float* __restrict__ br,
                                  const float* __restrict__ Wo,
                                  float* __restrict__ out) {
  int waveId = threadIdx.x >> 5;
  int lane   = threadIdx.x & 31;
  int tile   = blockIdx.x * (blockDim.x >> 5) + waveId;
  int mt = tile >> 3;
  int nt = tile & 7;
  int m     = mt * 16 + (lane & 15);
  int n     = nt * 16 + (lane & 15);
  int khalf = lane >> 4;

  float inv = 1.0f / fmaxf(cnt_a[m], 1.0f);
  const float* ra = sums_a   + (size_t)m * DIM;
  const float* rx = x_author + (size_t)m * DIM;
  const float* wr = Wr + (size_t)n * DIM;
  const float* wo = Wo + (size_t)n * DIM;

  v8f acc = {};
#pragma unroll
  for (int kb = 0; kb < 4; ++kb) {
    acc = WMMA_F16(make_a(ra, kb, khalf, inv),  make_b(wr, kb, khalf), acc);
    acc = WMMA_F16(make_a(rx, kb, khalf, 1.0f), make_b(wo, kb, khalf), acc);
  }

  float bias = br[n];
  int mrow = mt * 16 + khalf * 8;
#pragma unroll
  for (int r = 0; r < 8; ++r)
    out[(size_t)(mrow + r) * DIM + n] = acc[r] + bias;
}

// ----------------------------------------------------------------------------
extern "C" void kernel_launch(void* const* d_in, const int* in_sizes, int n_in,
                              void* d_out, int out_size, void* d_ws, size_t ws_size,
                              hipStream_t stream) {
  const float* x_paper   = (const float*)d_in[0];
  const float* x_author  = (const float*)d_in[1];
  const int*   wr_src    = (const int*)d_in[2];   // author -> paper
  const int*   wr_dst    = (const int*)d_in[3];
  const int*   wn_src    = (const int*)d_in[4];   // paper -> author
  const int*   wn_dst    = (const int*)d_in[5];
  const int*   ci_src    = (const int*)d_in[6];   // paper -> paper
  const int*   ci_dst    = (const int*)d_in[7];
  const float* Wr_writes  = (const float*)d_in[8];
  const float* br_writes  = (const float*)d_in[9];
  const float* Wo_writes  = (const float*)d_in[10];
  const float* Wr_written = (const float*)d_in[11];
  const float* br_written = (const float*)d_in[12];
  const float* Wo_written = (const float*)d_in[13];
  const float* Wr_cites   = (const float*)d_in[14];
  const float* br_cites   = (const float*)d_in[15];
  const float* Wo_cites   = (const float*)d_in[16];

  const int E_writes  = in_sizes[2];
  const int E_written = in_sizes[4];
  const int E_cites   = in_sizes[6];

  // Workspace layout (floats): 3 sum matrices + 3 float counts = ~62 MB.
  float* sums_w = (float*)d_ws;
  float* sums_c = sums_w + (size_t)NPAPER * DIM;
  float* sums_a = sums_c + (size_t)NPAPER * DIM;
  float* cnt_w  = sums_a + (size_t)NAUTHOR * DIM;
  float* cnt_c  = cnt_w + NPAPER;
  float* cnt_a  = cnt_c + NPAPER;
  size_t ws_floats = (size_t)NPAPER * DIM * 2 + (size_t)NAUTHOR * DIM
                   + (size_t)NPAPER * 2 + (size_t)NAUTHOR;
  hipMemsetAsync(d_ws, 0, ws_floats * sizeof(float), stream);

  // Phase 1: scatter-mean accumulation (one wave per edge, 8 edges/block).
  scatter_accum<<<(E_writes  + 7) / 8, 256, 0, stream>>>(x_author, wr_src, wr_dst, sums_w, cnt_w, E_writes);
  scatter_accum<<<(E_cites   + 7) / 8, 256, 0, stream>>>(x_paper,  ci_src, ci_dst, sums_c, cnt_c, E_cites);
  scatter_accum<<<(E_written + 7) / 8, 256, 0, stream>>>(x_paper,  wn_src, wn_dst, sums_a, cnt_a, E_written);

  float* out_paper  = (float*)d_out;
  float* out_author = out_paper + (size_t)NPAPER * DIM;

  // Phase 2: WMMA dense transforms. 8 waves/block, tile counts divide evenly.
  paper_out_kernel<<<(NPAPER / 16) * 8 / 8, 256, 0, stream>>>(
      sums_w, cnt_w, sums_c, cnt_c, x_paper,
      Wr_writes, br_writes, Wr_cites, br_cites, Wo_writes, Wo_cites, out_paper);
  author_out_kernel<<<(NAUTHOR / 16) * 8 / 8, 256, 0, stream>>>(
      sums_a, cnt_a, x_author, Wr_written, br_written, Wo_written, out_author);
}
